// PraxisAttention_43404939493769
// MI455X (gfx1250) — matrix operand order repaired and checked by reading, not compiled, not run on hardware
//
#include <hip/hip_runtime.h>
#include <hip/hip_bf16.h>

// Problem constants (reference: B,S,D,H = 2,2048,1024,16; HD=64)
#define BB   2
#define SS   2048
#define DDIM 1024
#define HH   16
#define HDD  64
#define MM   (BB * SS)      // 4096 rows for projection GEMMs
#define LDPAD 40            // LDS row stride (elements) for 32-wide tiles

typedef __attribute__((ext_vector_type(16))) __bf16 bf16x16;
typedef __attribute__((ext_vector_type(8)))  float  f32x8;

struct FragBF {
  union { bf16x16 v; __bf16 e[16]; uint4 q[2]; };
};

__device__ __forceinline__ f32x8 wmma_bf16(const FragBF& a, const FragBF& b, f32x8 c) {
  // D(16x16 f32) = A(16x32 bf16) * B(32x16 bf16) + C
  return __builtin_amdgcn_wmma_f32_16x16x32_bf16(false, a.v, false, b.v, (short)0, c,
                                                 false, false);
}

// ---- Fragment loaders (wave32 layouts per CDNA5 ISA 7.12.2) ----
// A 16x32: lane r(0..15)=row; half0 -> K[0..7]+K[16..23], half1 -> K[8..15]+K[24..31]
__device__ __forceinline__ FragBF load_frag_a(const __bf16* base, int ld, int lane) {
  const int r = lane & 15, hf = lane >> 4;
  const __bf16* p = base + (size_t)r * ld + hf * 8;
  FragBF f;
  f.q[0] = *(const uint4*)(p);
  f.q[1] = *(const uint4*)(p + 16);
  return f;
}
// B 32x16 from B^T rows ([N][K]): lane n=column; half0 -> K[0..7]+K[16..23], half1 -> K[8..15]+K[24..31]
__device__ __forceinline__ FragBF load_frag_b(const __bf16* base, int ld, int lane) {
  const int r = lane & 15, hf = lane >> 4;
  const __bf16* p = base + (size_t)r * ld + hf * 8;
  FragBF f;
  f.q[0] = *(const uint4*)(p);
  f.q[1] = *(const uint4*)(p + 16);
  return f;
}

// ---- CDNA5 async global->LDS staging (b128 per lane), ASYNCcnt tracked ----
__device__ __forceinline__ void async_b128(uint32_t lds_byte_off, const void* sbase,
                                           uint32_t vbyte_off) {
  asm volatile("global_load_async_to_lds_b128 %0, %1, %2"
               :: "v"(lds_byte_off), "v"(vbyte_off), "s"(sbase)
               : "memory");
}

// Stage one 128x32 bf16 tile (row stride ldg elements in global) into LDS
// (row stride LDPAD). 256 threads x 2 chunks of 16B each.
__device__ __forceinline__ void stage_tile_async(const __bf16* gbase, int ldg,
                                                 uint32_t lds_base, int tid) {
#pragma unroll
  for (int c = 0; c < 2; ++c) {
    const int chunk = tid * 2 + c;          // 0..511
    const int row = chunk >> 2;             // 0..127
    const int qt  = chunk & 3;              // 16B quarter of the 64B row
    const uint32_t voff = (uint32_t)((row * ldg + qt * 8) * 2);
    const uint32_t loff = lds_base + (uint32_t)((row * LDPAD + qt * 8) * 2);
    async_b128(loff, gbase, voff);
  }
}

// ---------------------------------------------------------------------------
// x (f32) -> bf16, vectorized
// ---------------------------------------------------------------------------
__global__ __launch_bounds__(256) void cvt_bf16_kernel(const float* __restrict__ in,
                                                       __bf16* __restrict__ out) {
  const size_t i = ((size_t)blockIdx.x * 256 + threadIdx.x) * 8;
  float4 a = ((const float4*)(in + i))[0];
  float4 b = ((const float4*)(in + i))[1];
  __align__(16) __bf16 t[8];
  t[0]=(__bf16)a.x; t[1]=(__bf16)a.y; t[2]=(__bf16)a.z; t[3]=(__bf16)a.w;
  t[4]=(__bf16)b.x; t[5]=(__bf16)b.y; t[6]=(__bf16)b.z; t[7]=(__bf16)b.w;
  *(uint4*)(out + i) = *(const uint4*)t;
}

// ---------------------------------------------------------------------------
// W f32 [K][N] -> W^T bf16 [N][K]  (64x64 tiles through LDS)
// ---------------------------------------------------------------------------
__global__ __launch_bounds__(256) void transpose_w_kernel(const float* __restrict__ W,
                                                          __bf16* __restrict__ Wt,
                                                          int N, int Kd) {
  __shared__ float sT[64][65];
  const int k0 = blockIdx.x * 64, n0 = blockIdx.y * 64;
  const int t = threadIdx.x;
  const int rr = t >> 2;            // 0..63
  const int cc = (t & 3) * 16;      // 0..48
  const float* src = W + (size_t)(k0 + rr) * N + n0 + cc;
#pragma unroll
  for (int i = 0; i < 4; ++i) {
    float4 f = ((const float4*)src)[i];
    sT[rr][cc + i * 4 + 0] = f.x; sT[rr][cc + i * 4 + 1] = f.y;
    sT[rr][cc + i * 4 + 2] = f.z; sT[rr][cc + i * 4 + 3] = f.w;
  }
  __syncthreads();
  __align__(16) __bf16 tmp[16];
#pragma unroll
  for (int i = 0; i < 16; ++i) tmp[i] = (__bf16)sT[cc + i][rr];
  __bf16* dst = Wt + (size_t)(n0 + rr) * Kd + k0 + cc;
  *(uint4*)(dst)     = *(const uint4*)(&tmp[0]);
  *(uint4*)(dst + 8) = *(const uint4*)(&tmp[8]);
}

// ---------------------------------------------------------------------------
// GEMM: C[M,N] = A[M,K](bf16) @ Bt[N,K]^T(bf16), async-LDS double-buffered.
// Block 256 thr = 8 waves (4x2), block tile 128x128, wave tile 32x64, BK=32.
// mode 0: f32 out + bias        (final projection)
// mode 1: bf16 out (acc+bias)*scale, [M][N]      (Q with 1/8, K)
// mode 2: bf16 out transposed [B,D,S], no bias   (V^T for attention PV)
// ---------------------------------------------------------------------------
__global__ __launch_bounds__(256) void gemm_kernel(
    const __bf16* __restrict__ A, const __bf16* __restrict__ Bt,
    const float* __restrict__ bias, void* __restrict__ Cout,
    int M, int N, int Kd, int mode, float scale) {
  __shared__ __align__(16) __bf16 sA[2][128 * LDPAD];
  __shared__ __align__(16) __bf16 sB[2][128 * LDPAD];

  const int tid  = threadIdx.x;
  const int wave = tid >> 5, lane = tid & 31;
  const int r = lane & 15, hf = lane >> 4;
  const int wm = wave >> 1, wn = wave & 1;
  const int bm = blockIdx.x * 128;
  const int bn = blockIdx.y * 128;

  const uint32_t sAoff[2] = { (uint32_t)(uintptr_t)&sA[0][0],
                              (uint32_t)(uintptr_t)&sA[1][0] };
  const uint32_t sBoff[2] = { (uint32_t)(uintptr_t)&sB[0][0],
                              (uint32_t)(uintptr_t)&sB[1][0] };

  f32x8 acc[2][4] = {};

  // preload first k-tile
  stage_tile_async(A  + (size_t)bm * Kd, Kd, sAoff[0], tid);
  stage_tile_async(Bt + (size_t)bn * Kd, Kd, sBoff[0], tid);

  int buf = 0;
  for (int k0 = 0; k0 < Kd; k0 += 32) {
    const bool has_next = (k0 + 32) < Kd;
    if (has_next) {
      stage_tile_async(A  + (size_t)bm * Kd + k0 + 32, Kd, sAoff[buf ^ 1], tid);
      stage_tile_async(Bt + (size_t)bn * Kd + k0 + 32, Kd, sBoff[buf ^ 1], tid);
      asm volatile("s_wait_asynccnt 0x4" ::: "memory");   // current buf done (in-order)
    } else {
      asm volatile("s_wait_asynccnt 0x0" ::: "memory");
    }
    __syncthreads();

    FragBF a0 = load_frag_a(&sA[buf][(wm * 32 + 0)  * LDPAD], LDPAD, lane);
    FragBF a1 = load_frag_a(&sA[buf][(wm * 32 + 16) * LDPAD], LDPAD, lane);
#pragma unroll
    for (int j = 0; j < 4; ++j) {
      FragBF bj = load_frag_b(&sB[buf][(wn * 64 + j * 16) * LDPAD], LDPAD, lane);
      acc[0][j] = wmma_bf16(a0, bj, acc[0][j]);
      acc[1][j] = wmma_bf16(a1, bj, acc[1][j]);
    }
    __syncthreads();
    buf ^= 1;
  }

  // ---- epilogue (C layout: lane -> column, VGPR idx -> row, +8 upper half) ----
  if (mode == 0) {
    float* Of = (float*)Cout;
#pragma unroll
    for (int i = 0; i < 2; ++i) {
#pragma unroll
      for (int j = 0; j < 4; ++j) {
        const int col = bn + wn * 64 + j * 16 + r;
        const float bv = bias ? bias[col] : 0.0f;
#pragma unroll
        for (int v2 = 0; v2 < 8; ++v2) {
          const int row = bm + wm * 32 + i * 16 + v2 + hf * 8;
          Of[(size_t)row * N + col] = acc[i][j][v2] + bv;
        }
      }
    }
  } else if (mode == 1) {
    __bf16* Ob = (__bf16*)Cout;
#pragma unroll
    for (int i = 0; i < 2; ++i) {
#pragma unroll
      for (int j = 0; j < 4; ++j) {
        const int col = bn + wn * 64 + j * 16 + r;
        const float bv = bias ? bias[col] : 0.0f;
#pragma unroll
        for (int v2 = 0; v2 < 8; ++v2) {
          const int row = bm + wm * 32 + i * 16 + v2 + hf * 8;
          Ob[(size_t)row * N + col] = (__bf16)((acc[i][j][v2] + bv) * scale);
        }
      }
    }
  } else {
    // mode 2: V^T bf16 [B, D, S]; 8 consecutive rows (s) pack into one b128
    __bf16* Ob = (__bf16*)Cout;
#pragma unroll
    for (int i = 0; i < 2; ++i) {
      const int row0 = bm + wm * 32 + i * 16 + hf * 8;   // rows row0..row0+7
      const int bb = row0 >> 11;                          // / SS
      const int ss0 = row0 & (SS - 1);
#pragma unroll
      for (int j = 0; j < 4; ++j) {
        const int col = bn + wn * 64 + j * 16 + r;        // d index
        __align__(16) __bf16 tmp[8];
#pragma unroll
        for (int v2 = 0; v2 < 8; ++v2) tmp[v2] = (__bf16)acc[i][j][v2];
        *(uint4*)(Ob + ((size_t)bb * DDIM + col) * SS + ss0) = *(const uint4*)tmp;
      }
    }
  }
}

// ---------------------------------------------------------------------------
// Causal flash attention, fully register-resident (no LDS).
// Q,K bf16 [B,S,D] (Q pre-scaled by 1/8), Vt bf16 [B,D,S].
// Trick: compute S^T = K * Q^T. The C-layout of S^T (lane = query column,
// VGPR index = key row) is bit-identical to the WMMA A-fragment layout of P
// for the PV matmul, so probabilities never leave VGPRs. Each lane owns one
// query: softmax stats are per-lane scalars (in-lane reduction + one
// shfl_xor(16)); accumulator rescale uses 8 shfl broadcasts.
// ALiBi key bias -slope_h*j; -1e9 on j>i (reference soft-mask mean; exp
// underflows to 0 in fp32 exactly like the reference).
// ---------------------------------------------------------------------------
__global__ __launch_bounds__(256) void attn_kernel(
    const __bf16* __restrict__ Qm, const __bf16* __restrict__ Km,
    const __bf16* __restrict__ Vt, __bf16* __restrict__ Om) {
  const int tid  = threadIdx.x;
  const int wave = tid >> 5, lane = tid & 31;
  const int r = lane & 15, hf = lane >> 4;

  const int tileId = blockIdx.x * 8 + wave;
  const int bh = tileId >> 7;              // / (SS/16 = 128)
  const int it = tileId & 127;
  const int b  = bh >> 4;                  // / HH
  const int h  = bh & 15;
  const int i0 = it * 16;

  const size_t rowBase = (size_t)b * SS;
  const __bf16* Qb = Qm + (rowBase + i0) * DDIM + h * HDD;
  const __bf16* Kb = Km + rowBase * DDIM + h * HDD;
  const __bf16* Vb = Vt + ((size_t)b * DDIM + h * HDD) * SS;

  const float slope = exp2f(-0.5f * (float)(h + 1));   // 2^(-8*(h+1)/16)

  // Q^T as B-fragments (lane = query column, half0 hd 0..7+16..23, half1 hd 8..15+24..31)
  FragBF qtf0 = load_frag_b(Qb,      DDIM, lane);   // hd 0..31
  FragBF qtf1 = load_frag_b(Qb + 32, DDIM, lane);   // hd 32..63

  const int qrow = i0 + r;      // this lane's query (same for lane and lane^16)
  float m = -3.0e38f, l = 0.0f;
  f32x8 o0 = {}, o1 = {}, o2 = {}, o3 = {};

  const int jmax = i0 + 15;
  for (int j0 = 0; j0 <= jmax; j0 += 64) {
    // ---- S^T = K*Q^T for 64 keys: 4 key tiles x 2 hd K-steps = 8 WMMA ----
    f32x8 s0 = {}, s1 = {}, s2 = {}, s3 = {};
    {
      FragBF ka;
      ka = load_frag_a(Kb + (size_t)(j0 +  0) * DDIM,      DDIM, lane); s0 = wmma_bf16(ka, qtf0, s0);
      ka = load_frag_a(Kb + (size_t)(j0 +  0) * DDIM + 32, DDIM, lane); s0 = wmma_bf16(ka, qtf1, s0);
      ka = load_frag_a(Kb + (size_t)(j0 + 16) * DDIM,      DDIM, lane); s1 = wmma_bf16(ka, qtf0, s1);
      ka = load_frag_a(Kb + (size_t)(j0 + 16) * DDIM + 32, DDIM, lane); s1 = wmma_bf16(ka, qtf1, s1);
      ka = load_frag_a(Kb + (size_t)(j0 + 32) * DDIM,      DDIM, lane); s2 = wmma_bf16(ka, qtf0, s2);
      ka = load_frag_a(Kb + (size_t)(j0 + 32) * DDIM + 32, DDIM, lane); s2 = wmma_bf16(ka, qtf1, s2);
      ka = load_frag_a(Kb + (size_t)(j0 + 48) * DDIM,      DDIM, lane); s3 = wmma_bf16(ka, qtf0, s3);
      ka = load_frag_a(Kb + (size_t)(j0 + 48) * DDIM + 32, DDIM, lane); s3 = wmma_bf16(ka, qtf1, s3);
    }

    // ---- bias + causal mask, in-lane (this lane holds 32 of 64 keys) ----
    float mx = -3.0e38f;
#pragma unroll
    for (int v2 = 0; v2 < 8; ++v2) {
      const int jb = j0 + v2 + hf * 8;     // key for tile 0
      const float x0 = s0[v2] - slope * (float)(jb +  0) + ((jb +  0 > qrow) ? -1.0e9f : 0.0f);
      const float x1 = s1[v2] - slope * (float)(jb + 16) + ((jb + 16 > qrow) ? -1.0e9f : 0.0f);
      const float x2 = s2[v2] - slope * (float)(jb + 32) + ((jb + 32 > qrow) ? -1.0e9f : 0.0f);
      const float x3 = s3[v2] - slope * (float)(jb + 48) + ((jb + 48 > qrow) ? -1.0e9f : 0.0f);
      s0[v2] = x0; s1[v2] = x1; s2[v2] = x2; s3[v2] = x3;
      mx = fmaxf(mx, fmaxf(fmaxf(x0, x1), fmaxf(x2, x3)));
    }
    mx = fmaxf(mx, __shfl_xor(mx, 16));    // combine with partner half
    const float mn = fmaxf(m, mx);

    // ---- exp + row sum; repack probabilities as the PV A-fragments ----
    FragBF pA0, pA1;
    float ps = 0.0f;
#pragma unroll
    for (int v2 = 0; v2 < 8; ++v2) {
      const float p0 = __expf(s0[v2] - mn);
      const float p1 = __expf(s1[v2] - mn);
      const float p2 = __expf(s2[v2] - mn);
      const float p3 = __expf(s3[v2] - mn);
      ps += (p0 + p1) + (p2 + p3);
      // A-frag: e[i] = K(hf*8+i), e[8+i] = K(16+hf*8+i) -> exactly tiles 0/1 (2/3)
      pA0.e[v2]     = (__bf16)p0;
      pA0.e[8 + v2] = (__bf16)p1;
      pA1.e[v2]     = (__bf16)p2;
      pA1.e[8 + v2] = (__bf16)p3;
    }
    ps += __shfl_xor(ps, 16);

    const float alpha = __expf(m - mn);
    m = mn;
    l = l * alpha + ps;

    // broadcast per-query alpha to the hd lanes of each output row
#pragma unroll
    for (int v2 = 0; v2 < 8; ++v2) {
      const float av = __shfl(alpha, v2 + hf * 8);
      o0[v2] *= av; o1[v2] *= av; o2[v2] *= av; o3[v2] *= av;
    }

    // ---- PV: O(16q x 64hd) += P(16q x 64k) @ V(64k x 64hd), 8 WMMA ----
    {
      FragBF vf;
      const __bf16* v0 = Vb + (size_t)(0 * 16) * SS + j0;
      vf = load_frag_b(v0,      SS, lane); o0 = wmma_bf16(pA0, vf, o0);
      vf = load_frag_b(v0 + 32, SS, lane); o0 = wmma_bf16(pA1, vf, o0);
      const __bf16* v1 = Vb + (size_t)(1 * 16) * SS + j0;
      vf = load_frag_b(v1,      SS, lane); o1 = wmma_bf16(pA0, vf, o1);
      vf = load_frag_b(v1 + 32, SS, lane); o1 = wmma_bf16(pA1, vf, o1);
      const __bf16* v2p = Vb + (size_t)(2 * 16) * SS + j0;
      vf = load_frag_b(v2p,      SS, lane); o2 = wmma_bf16(pA0, vf, o2);
      vf = load_frag_b(v2p + 32, SS, lane); o2 = wmma_bf16(pA1, vf, o2);
      const __bf16* v3 = Vb + (size_t)(3 * 16) * SS + j0;
      vf = load_frag_b(v3,      SS, lane); o3 = wmma_bf16(pA0, vf, o3);
      vf = load_frag_b(v3 + 32, SS, lane); o3 = wmma_bf16(pA1, vf, o3);
    }
  }

  // ---- normalize (broadcast per-query 1/l), write bf16 [B,S,D] ----
  const float linv = 1.0f / l;
#pragma unroll
  for (int v2 = 0; v2 < 8; ++v2) {
    const float lv = __shfl(linv, v2 + hf * 8);
    const int row = i0 + v2 + hf * 8;
    __bf16* op = Om + (rowBase + row) * DDIM + h * HDD;
    op[0 * 16 + r] = (__bf16)(o0[v2] * lv);
    op[1 * 16 + r] = (__bf16)(o1[v2] * lv);
    op[2 * 16 + r] = (__bf16)(o2[v2] * lv);
    op[3 * 16 + r] = (__bf16)(o3[v2] * lv);
  }
}

// ---------------------------------------------------------------------------
extern "C" void kernel_launch(void* const* d_in, const int* in_sizes, int n_in,
                              void* d_out, int out_size, void* d_ws, size_t ws_size,
                              hipStream_t stream) {
  const float* x  = (const float*)d_in[0];
  const float* Wq = (const float*)d_in[1];
  const float* bq = (const float*)d_in[2];
  const float* Wk = (const float*)d_in[3];
  const float* bk = (const float*)d_in[4];
  const float* Wv = (const float*)d_in[5];
  const float* Wo = (const float*)d_in[6];
  const float* bo = (const float*)d_in[7];
  float* out = (float*)d_out;

  const size_t MD = (size_t)MM * DDIM;        // 4096*1024
  const size_t DD2 = (size_t)DDIM * DDIM;     // 1024*1024
  __bf16* xb  = (__bf16*)d_ws;
  __bf16* Wqt = xb  + MD;
  __bf16* Wkt = Wqt + DD2;
  __bf16* Wvt = Wkt + DD2;
  __bf16* Wot = Wvt + DD2;
  __bf16* Qb  = Wot + DD2;
  __bf16* Kb  = Qb  + MD;
  __bf16* Vtb = Kb  + MD;
  __bf16* AOb = Vtb + MD;

  // prep: x -> bf16, weights -> transposed bf16
  cvt_bf16_kernel<<<dim3(MD / (256 * 8)), 256, 0, stream>>>(x, xb);
  dim3 tg(DDIM / 64, DDIM / 64);
  transpose_w_kernel<<<tg, 256, 0, stream>>>(Wq, Wqt, DDIM, DDIM);
  transpose_w_kernel<<<tg, 256, 0, stream>>>(Wk, Wkt, DDIM, DDIM);
  transpose_w_kernel<<<tg, 256, 0, stream>>>(Wv, Wvt, DDIM, DDIM);
  transpose_w_kernel<<<tg, 256, 0, stream>>>(Wo, Wot, DDIM, DDIM);

  // projections
  dim3 gg(MM / 128, DDIM / 128);   // (32, 8)
  gemm_kernel<<<gg, 256, 0, stream>>>(xb, Wqt, bq, Qb,  MM, DDIM, DDIM, 1, 0.125f);
  gemm_kernel<<<gg, 256, 0, stream>>>(xb, Wkt, bk, Kb,  MM, DDIM, DDIM, 1, 1.0f);
  gemm_kernel<<<gg, 256, 0, stream>>>(xb, Wvt, nullptr, Vtb, MM, DDIM, DDIM, 2, 1.0f);

  // attention (no LDS, fully register-resident P)
  const int nTiles = BB * HH * (SS / 16);      // 4096 tiles, 8 per block
  attn_kernel<<<dim3(nTiles / 8), 256, 0, stream>>>(Qb, Kb, Vtb, AOb);

  // output projection (f32 out + bias)
  gemm_kernel<<<gg, 256, 0, stream>>>(AOb, Wot, bo, out, MM, DDIM, DDIM, 0, 1.0f);
}
